// RGCNLitmusEmbedder_18906446037169
// MI455X (gfx1250) — matrix operand order repaired and checked by reading, not compiled, not run on hardware
//
#include <hip/hip_runtime.h>
#include <hip/hip_bf16.h>

// ---------------------------------------------------------------------------
// Problem constants (match reference)
// ---------------------------------------------------------------------------
constexpr int NN   = 50000;
constexpr int EE   = 800000;
constexpr int FIN  = 128;
constexpr int RR   = 8;
constexpr int BB   = 64;
constexpr int H1   = 64;
constexpr int H2   = 128;
constexpr int EMBD = 16;
constexpr int OUTD = 32;
constexpr int NC1  = RR * H1 + H1;   // 576  : 8 relation blocks (64) + root block (64)
constexpr int NC2  = RR * H2 + H2;   // 1152 : 8 relation blocks (128) + root block (128)

typedef __attribute__((ext_vector_type(16))) __bf16          v16bf;
typedef __attribute__((ext_vector_type(8)))  float           v8f;
typedef __attribute__((ext_vector_type(8)))  unsigned short  v8us;

union FragU { v8us u[2]; v16bf v; };

// ---------------------------------------------------------------------------
// Utility kernels
// ---------------------------------------------------------------------------
__global__ void zero_kernel(unsigned int* __restrict__ p, long n) {
  long i = (long)blockIdx.x * blockDim.x + threadIdx.x;
  if (i < n) p[i] = 0u;
}

__global__ void cvt_bf16_kernel(const float* __restrict__ in,
                                __hip_bfloat16* __restrict__ out, long n) {
  long i = (long)blockIdx.x * blockDim.x + threadIdx.x;
  if (i < n) out[i] = __float2bfloat16(in[i]);
}

// Bt1[col][k] (bf16, [NC1][FIN]) from W1 [R][FIN][H1] and root1 [FIN][H1]
__global__ void build_bt1_kernel(const float* __restrict__ W1,
                                 const float* __restrict__ root1,
                                 __hip_bfloat16* __restrict__ Bt) {
  int idx = blockIdx.x * blockDim.x + threadIdx.x;
  if (idx >= NC1 * FIN) return;
  int col = idx / FIN;
  int k   = idx - col * FIN;
  float v;
  if (col < RR * H1) {
    int r = col >> 6, e = col & 63;
    v = W1[((size_t)r * FIN + k) * H1 + e];
  } else {
    v = root1[(size_t)k * H1 + (col - RR * H1)];
  }
  Bt[(size_t)col * FIN + k] = __float2bfloat16(v);
}

// Bt2[col][k] (bf16, [NC2][H1]) from W2 [R][H1][H2] and root2 [H1][H2]
__global__ void build_bt2_kernel(const float* __restrict__ W2,
                                 const float* __restrict__ root2,
                                 __hip_bfloat16* __restrict__ Bt) {
  int idx = blockIdx.x * blockDim.x + threadIdx.x;
  if (idx >= NC2 * H1) return;
  int col = idx / H1;
  int k   = idx - col * H1;
  float v;
  if (col < RR * H2) {
    int r = col >> 7, e = col & 127;
    v = W2[((size_t)r * H1 + k) * H2 + e];
  } else {
    v = root2[(size_t)k * H2 + (col - RR * H2)];
  }
  Bt[(size_t)col * H1 + k] = __float2bfloat16(v);
}

// ---------------------------------------------------------------------------
// WMMA GEMM: C[M,Ncols] = A[M,K](bf16 row-major) * Bt[Ncols,K](bf16, i.e. B^T)
// One wave -> one 16x16 tile of C; fp32 accumulation via v_wmma_f32_16x16x32_bf16.
// blockDim = 128 (4 waves -> 4 adjacent N-tiles), grid = (M/16, Ncols/64).
// ---------------------------------------------------------------------------
__global__ void wmma_gemm_bf16_kernel(const unsigned short* __restrict__ A,
                                      const unsigned short* __restrict__ Bt,
                                      float* __restrict__ C,
                                      int M, int K, int Ncols) {
  const int wave = threadIdx.x >> 5;
  const int lane = threadIdx.x & 31;
  const int m0 = blockIdx.x * 16;
  const int n0 = (blockIdx.y * 4 + wave) * 16;
  if (n0 >= Ncols || m0 >= M) return;

  const int kh  = lane >> 4;   // half-wave select
  const int l15 = lane & 15;

  // A frag: row m0+l15, K at kh*8 and 16+kh*8  (ISA 16-bit A 16x32 layout)
  const unsigned short* arow = A + (size_t)(m0 + l15) * K + kh * 8;
  // B frag: col n0+l15, K at kh*8 and 16+kh*8 (B stored transposed)
  const unsigned short* brow = Bt + (size_t)(n0 + l15) * K + kh * 8;

  v8f acc = {};
  for (int kb = 0; kb < K; kb += 32) {
    FragU af, bf;
    af.u[0] = *(const v8us*)(arow + kb);        // K = kh*8 + 0..7
    af.u[1] = *(const v8us*)(arow + kb + 16);   // K = 16 + kh*8 + 0..7
    bf.u[0] = *(const v8us*)(brow + kb);        // K = kh*8 + 0..7
    bf.u[1] = *(const v8us*)(brow + kb + 16);   // K = 16 + kh*8 + 0..7
    acc = __builtin_amdgcn_wmma_f32_16x16x32_bf16(
        /*neg_a=*/false, af.v, /*neg_b=*/false, bf.v,
        /*c_mod=*/(short)0, acc, /*reuse_a=*/false, /*reuse_b=*/false);
  }

  // D layout: lane l -> col l&15, rows m0 + 8*(l>>4) + v
  float* crow = C + (size_t)(m0 + 8 * kh) * Ncols + n0 + l15;
#pragma unroll
  for (int v = 0; v < 8; ++v) crow[(size_t)v * Ncols] = acc[v];
}

// ---------------------------------------------------------------------------
// Edge (dst, relation) degree counts
// ---------------------------------------------------------------------------
__global__ void count_edges_kernel(const int* __restrict__ dst,
                                   const int* __restrict__ et,
                                   int* __restrict__ cnt) {
  int e = blockIdx.x * blockDim.x + threadIdx.x;
  if (e < EE) atomicAdd(&cnt[dst[e] * RR + et[e]], 1);
}

// agg[n][j] = Ctab[n][off + j] + bias[j]    (root term + bias, pre-scatter init)
__global__ void init_agg_kernel(const float* __restrict__ Ctab,
                                const float* __restrict__ bias,
                                float* __restrict__ agg,
                                int ld, int off, int logW, long total) {
  long idx = (long)blockIdx.x * blockDim.x + threadIdx.x;
  if (idx >= total) return;
  int n = (int)(idx >> logW);
  int j = (int)idx & ((1 << logW) - 1);
  agg[idx] = Ctab[(size_t)n * ld + off + j] + bias[j];
}

// Per edge e, feature j: agg[dst] += Ctab[src][rel_block + j] / cnt[dst,rel]
__global__ void scatter_edges_kernel(const float* __restrict__ Ctab,
                                     const int* __restrict__ src,
                                     const int* __restrict__ dst,
                                     const int* __restrict__ et,
                                     const int* __restrict__ cnt,
                                     float* __restrict__ agg,
                                     int ld, int logW, long total) {
  long t = (long)blockIdx.x * blockDim.x + threadIdx.x;
  if (t >= total) return;
  int e = (int)(t >> logW);
  int j = (int)t & ((1 << logW) - 1);
  int s = src[e], d = dst[e], r = et[e];
  int c = cnt[d * RR + r];
  float norm = 1.0f / (float)(c > 0 ? c : 1);
  const float* mrow = Ctab + (size_t)s * ld + (r << logW);
  __builtin_prefetch(mrow, 0, 1);   // global_prefetch_b8
  atomicAdd(&agg[((size_t)d << logW) + j], mrow[j] * norm);
}

// ReLU + convert to bf16 for the next WMMA GEMM
__global__ void relu_bf16_kernel(const float* __restrict__ agg,
                                 __hip_bfloat16* __restrict__ out, long n) {
  long i = (long)blockIdx.x * blockDim.x + threadIdx.x;
  if (i >= n) return;
  float h = agg[i];
  out[i] = __float2bfloat16(h > 0.0f ? h : 0.0f);
}

// ReLU layer-2 output + dual pooling (mean-sum / max, per-graph counts)
__global__ void relu_pool_kernel(const float* __restrict__ agg2,
                                 const int* __restrict__ batch,
                                 float* __restrict__ meansum,
                                 float* __restrict__ maxp,
                                 int* __restrict__ cntb) {
  long idx = (long)blockIdx.x * blockDim.x + threadIdx.x;
  if (idx >= (long)NN * H2) return;
  int n = (int)(idx >> 7);
  int j = (int)idx & 127;
  float h = agg2[idx];
  h = h > 0.0f ? h : 0.0f;
  int b = batch[n];
  atomicAdd(&meansum[b * H2 + j], h);
  // h >= 0 so integer compare of IEEE bits == float compare; maxp init 0.0f
  atomicMax((int*)&maxp[b * H2 + j], __float_as_int(h));
  if (j == 0) atomicAdd(&cntb[b], 1);
}

// ---------------------------------------------------------------------------
// Head: g = mean + max -> fc1 -> BN(batch stats) -> LeakyReLU -> fc2 (emb)
//       -> dec1 -> LeakyReLU -> dec2 (pred).  Single block, LDS staged.
// out[0..2047] = pred [64,32]; out[2048..3071] = emb [64,16]
// ---------------------------------------------------------------------------
__global__ void head_kernel(const float* __restrict__ meansum,
                            const float* __restrict__ maxp,
                            const int* __restrict__ cntb,
                            const float* __restrict__ fc1_w, const float* __restrict__ fc1_b,
                            const float* __restrict__ bn_g,  const float* __restrict__ bn_b,
                            const float* __restrict__ fc2_w, const float* __restrict__ fc2_b,
                            const float* __restrict__ dec1_w, const float* __restrict__ dec1_b,
                            const float* __restrict__ dec2_w, const float* __restrict__ dec2_b,
                            float* __restrict__ out) {
  __shared__ float g[BB][H2];     // 32 KB
  __shared__ float z[BB][H1];     // 16 KB (reused later for decoder hidden)
  __shared__ float emb[BB][EMBD]; //  4 KB
  __shared__ float mu[H1], rstd[H1];
  const int tid = threadIdx.x;    // 256 threads

  for (int idx = tid; idx < BB * H2; idx += 256) {
    int i = idx >> 7;
    int c = cntb[i];
    float cf = (float)(c > 0 ? c : 1);
    g[i][idx & 127] = meansum[idx] / cf + maxp[idx];
  }
  __syncthreads();

  for (int idx = tid; idx < BB * H1; idx += 256) {
    int i = idx >> 6, o = idx & 63;
    float s = fc1_b[o];
    for (int k = 0; k < H2; ++k) s += g[i][k] * fc1_w[k * H1 + o];
    z[i][o] = s;
  }
  __syncthreads();

  if (tid < H1) {
    float m = 0.0f;
    for (int i = 0; i < BB; ++i) m += z[i][tid];
    m *= (1.0f / BB);
    float v = 0.0f;
    for (int i = 0; i < BB; ++i) { float d = z[i][tid] - m; v += d * d; }
    v *= (1.0f / BB);
    mu[tid] = m;
    rstd[tid] = rsqrtf(v + 1e-5f);
  }
  __syncthreads();

  for (int idx = tid; idx < BB * H1; idx += 256) {
    int i = idx >> 6, o = idx & 63;
    float zz = (z[i][o] - mu[o]) * rstd[o] * bn_g[o] + bn_b[o];
    z[i][o] = zz > 0.0f ? zz : 0.2f * zz;
  }
  __syncthreads();

  for (int idx = tid; idx < BB * EMBD; idx += 256) {
    int i = idx >> 4, o = idx & 15;
    float s = fc2_b[o];
    for (int k = 0; k < H1; ++k) s += z[i][k] * fc2_w[k * EMBD + o];
    emb[i][o] = s;
    out[BB * OUTD + idx] = s;          // emb output
  }
  __syncthreads();

  for (int idx = tid; idx < BB * H1; idx += 256) {  // reuse z as decoder hidden
    int i = idx >> 6, o = idx & 63;
    float s = dec1_b[o];
    for (int k = 0; k < EMBD; ++k) s += emb[i][k] * dec1_w[k * H1 + o];
    z[i][o] = s > 0.0f ? s : 0.2f * s;
  }
  __syncthreads();

  for (int idx = tid; idx < BB * OUTD; idx += 256) {
    int i = idx >> 5, o = idx & 31;
    float s = dec2_b[o];
    for (int k = 0; k < H1; ++k) s += z[i][k] * dec2_w[k * OUTD + o];
    out[idx] = s;                      // pred output
  }
}

// ---------------------------------------------------------------------------
// Host launch
// ---------------------------------------------------------------------------
extern "C" void kernel_launch(void* const* d_in, const int* in_sizes, int n_in,
                              void* d_out, int out_size, void* d_ws, size_t ws_size,
                              hipStream_t stream) {
  const float* x      = (const float*)d_in[0];
  const int*   ei     = (const int*)d_in[1];
  const int*   src    = ei;
  const int*   dstp   = ei + EE;
  const int*   et     = (const int*)d_in[2];
  const int*   batch  = (const int*)d_in[3];
  const float* W1     = (const float*)d_in[4];
  const float* root1  = (const float*)d_in[5];
  const float* b1     = (const float*)d_in[6];
  const float* W2     = (const float*)d_in[7];
  const float* root2  = (const float*)d_in[8];
  const float* b2     = (const float*)d_in[9];
  const float* fc1_w  = (const float*)d_in[10];
  const float* fc1_b  = (const float*)d_in[11];
  const float* bn_g   = (const float*)d_in[12];
  const float* bn_b   = (const float*)d_in[13];
  const float* fc2_w  = (const float*)d_in[14];
  const float* fc2_b  = (const float*)d_in[15];
  const float* dec1_w = (const float*)d_in[16];
  const float* dec1_b = (const float*)d_in[17];
  const float* dec2_w = (const float*)d_in[18];
  const float* dec2_b = (const float*)d_in[19];
  float* out = (float*)d_out;

  // Workspace carve (256B aligned slices)
  char* p = (char*)d_ws;
  auto carve = [&p](size_t bytes) -> void* {
    void* r = (void*)p;
    p += (bytes + 255) & ~(size_t)255;
    return r;
  };
  __hip_bfloat16* xbf   = (__hip_bfloat16*)carve((size_t)NN * FIN * 2);
  __hip_bfloat16* Bt1   = (__hip_bfloat16*)carve((size_t)NC1 * FIN * 2);
  float*          C1    = (float*)carve((size_t)NN * NC1 * 4);
  int*            cnt   = (int*)carve((size_t)NN * RR * 4);
  float*          agg1  = (float*)carve((size_t)NN * H1 * 4);
  __hip_bfloat16* h1bf  = (__hip_bfloat16*)carve((size_t)NN * H1 * 2);
  __hip_bfloat16* Bt2   = (__hip_bfloat16*)carve((size_t)NC2 * H1 * 2);
  float*          C2    = (float*)carve((size_t)NN * NC2 * 4);
  float*          agg2  = (float*)carve((size_t)NN * H2 * 4);
  float*          msum  = (float*)carve((size_t)BB * H2 * 4);
  float*          maxp  = (float*)carve((size_t)BB * H2 * 4);
  int*            cntb  = (int*)carve((size_t)BB * 4);

  const int T = 256;
  auto blocks = [](long n, int t) { return (unsigned)((n + t - 1) / t); };

  // Zero accumulation state (atomics accumulate -> must reset every call)
  zero_kernel<<<blocks((long)NN * RR, T), T, 0, stream>>>((unsigned*)cnt, (long)NN * RR);
  zero_kernel<<<blocks((long)BB * H2, T), T, 0, stream>>>((unsigned*)msum, (long)BB * H2);
  zero_kernel<<<blocks((long)BB * H2, T), T, 0, stream>>>((unsigned*)maxp, (long)BB * H2);
  zero_kernel<<<1, T, 0, stream>>>((unsigned*)cntb, (long)BB);

  // Prep: bf16 inputs + transposed bf16 weight tables
  cvt_bf16_kernel<<<blocks((long)NN * FIN, T), T, 0, stream>>>(x, xbf, (long)NN * FIN);
  build_bt1_kernel<<<blocks((long)NC1 * FIN, T), T, 0, stream>>>(W1, root1, Bt1);
  build_bt2_kernel<<<blocks((long)NC2 * H1, T), T, 0, stream>>>(W2, root2, Bt2);

  // Layer 1: per-relation transforms + root fused into one WMMA GEMM
  {
    dim3 grid(NN / 16, NC1 / 64);
    wmma_gemm_bf16_kernel<<<grid, 128, 0, stream>>>(
        (const unsigned short*)xbf, (const unsigned short*)Bt1, C1, NN, FIN, NC1);
  }
  count_edges_kernel<<<blocks(EE, T), T, 0, stream>>>(dstp, et, cnt);
  init_agg_kernel<<<blocks((long)NN * H1, T), T, 0, stream>>>(
      C1, b1, agg1, NC1, RR * H1, 6, (long)NN * H1);
  scatter_edges_kernel<<<blocks((long)EE * H1, T), T, 0, stream>>>(
      C1, src, dstp, et, cnt, agg1, NC1, 6, (long)EE * H1);
  relu_bf16_kernel<<<blocks((long)NN * H1, T), T, 0, stream>>>(agg1, h1bf, (long)NN * H1);

  // Layer 2
  {
    dim3 grid(NN / 16, NC2 / 64);
    wmma_gemm_bf16_kernel<<<grid, 128, 0, stream>>>(
        (const unsigned short*)h1bf, (const unsigned short*)Bt2, C2, NN, H1, NC2);
  }
  init_agg_kernel<<<blocks((long)NN * H2, T), T, 0, stream>>>(
      C2, b2, agg2, NC2, RR * H2, 7, (long)NN * H2);
  scatter_edges_kernel<<<blocks((long)EE * H2, T), T, 0, stream>>>(
      C2, src, dstp, et, cnt, agg2, NC2, 7, (long)EE * H2);

  // ReLU + dual pooling
  relu_pool_kernel<<<blocks((long)NN * H2, T), T, 0, stream>>>(agg2, batch, msum, maxp, cntb);

  // Head MLP (single block)
  head_kernel<<<1, 256, 0, stream>>>(msum, maxp, cntb,
                                     fc1_w, fc1_b, bn_g, bn_b, fc2_w, fc2_b,
                                     dec1_w, dec1_b, dec2_w, dec2_b, out);
}
